// EfficientMixedScoreMultiHeadAttentionLayer_70068096467029
// MI455X (gfx1250) — matrix operand order repaired and checked by reading, not compiled, not run on hardware
//
#include <hip/hip_runtime.h>
#include <hip/hip_bf16.h>

// ---------------------------------------------------------------------------
// EfficientMixedScoreMultiHeadAttentionLayer for MI455X (gfx1250, wave32)
// All GEMM-shaped work routed through v_wmma_f32_16x16x32_f16.
// B=8, R=C=200, E=256, H=16, D=16, MS_HID=32, SDIMS=2
// Padded dims: RP=208 (13*16 rows), CP=224 (7*32 K for combine GEMMs)
// ---------------------------------------------------------------------------

typedef _Float16 v16h __attribute__((ext_vector_type(16)));
typedef _Float16 v8h  __attribute__((ext_vector_type(8)));
typedef float    v8f  __attribute__((ext_vector_type(8)));

union H16 { v16h v; v8h h[2]; };
union FU { float f; unsigned int u; };

__device__ __forceinline__ v8f wmma_f16(v16h a, v16h b, v8f c) {
  // D = A(16x32,f16) * B(32x16,f16) + C(16x16,f32)
  return __builtin_amdgcn_wmma_f32_16x16x32_f16(
      /*neg_a=*/false, a, /*neg_b=*/false, b,
      /*c_mod=*/(short)0, c, /*reuse_a=*/false, /*reuse_b=*/false);
}

// A-operand tile (16x32 f16) from row-major storage, stride in halfs.
// ISA layout: lanes 0-15 M=0-15 hold K{0..7,16..23}; lanes 16-31 hold K{8..15,24..31}.
__device__ __forceinline__ v16h load_a_tile(const _Float16* base, int stride, int lane) {
  int row = lane & 15;
  int k1  = (lane >= 16) ? 8 : 0;
  const _Float16* p = base + row * stride;
  H16 u;
  u.h[0] = *(const v8h*)(p + k1);
  u.h[1] = *(const v8h*)(p + 16 + k1);
  return u.v;
}

// B-operand tile (32x16) from W stored (N,K) row-major (i.e. B^T), stride in halfs.
// ISA layout: lanes 0-15 N=0-15 hold K{0..7,16..23}; lanes 16-31 hold K{8..15,24..31}.
__device__ __forceinline__ v16h load_b_tile(const _Float16* base, int stride, int lane) {
  int n = lane & 15;
  const _Float16* p = base + n * stride + ((lane >= 16) ? 8 : 0);
  H16 u;
  u.h[0] = *(const v8h*)(p);
  u.h[1] = *(const v8h*)(p + 16);
  return u.v;
}

__device__ __forceinline__ unsigned int pack2h(_Float16 a, _Float16 b) {
  union { _Float16 h[2]; unsigned int u; } x;
  x.h[0] = a; x.h[1] = b;
  return x.u;
}

// ---------------------------------------------------------------------------
// f32 -> f16 weight conversion
// ---------------------------------------------------------------------------
__global__ void cvt_f16_kernel(const float* __restrict__ src,
                               _Float16* __restrict__ dst, int n) {
  int i = blockIdx.x * 256 + threadIdx.x;
  if (i < n) dst[i] = (_Float16)src[i];
}

// ---------------------------------------------------------------------------
// Generic GEMM: C(MxN,f32) = A(MxK,f32) @ W(NxK,f16)^T
// Requires M % 64 == 0, N % 64 == 0, K % 32 == 0 (true for all call sites).
// Block: 256 thr (8 waves). Tile 64x64. A staged/converted to f16 in LDS
// with unconditional b128 loads (no per-element exec masking).
// ---------------------------------------------------------------------------
__global__ void __launch_bounds__(256) gemm_f16w_kernel(
    const float* __restrict__ A, const _Float16* __restrict__ W,
    float* __restrict__ Cout, int M, int N, int K) {
  __shared__ _Float16 As[64 * 32] __attribute__((aligned(16)));
  int bm = blockIdx.x, bn = blockIdx.y;
  int t = threadIdx.x, lane = t & 31, w = t >> 5;
  int msub  = w & 3;          // 0..3  (m subtile)
  int nbase = (w >> 2) * 2;   // 0 or 2 (two n subtiles per wave)
  int r  = t >> 2;            // staging row 0..63
  int cc = (t & 3) * 8;       // staging col 0,8,16,24
  const float* arow = A + (size_t)(bm * 64 + r) * K + cc;
  v8f acc0 = {}, acc1 = {};
  for (int k0 = 0; k0 < K; k0 += 32) {
    float4 f0 = *(const float4*)(arow + k0);
    float4 f1 = *(const float4*)(arow + k0 + 4);
    v8h tmp;
    tmp[0] = (_Float16)f0.x; tmp[1] = (_Float16)f0.y;
    tmp[2] = (_Float16)f0.z; tmp[3] = (_Float16)f0.w;
    tmp[4] = (_Float16)f1.x; tmp[5] = (_Float16)f1.y;
    tmp[6] = (_Float16)f1.z; tmp[7] = (_Float16)f1.w;
    *(v8h*)(As + r * 32 + cc) = tmp;
    __syncthreads();
    v16h a  = load_a_tile(As + msub * 16 * 32, 32, lane);
    v16h b0 = load_b_tile(W + (size_t)(bn * 64 + (nbase + 0) * 16) * K + k0, K, lane);
    v16h b1 = load_b_tile(W + (size_t)(bn * 64 + (nbase + 1) * 16) * K + k0, K, lane);
    acc0 = wmma_f16(a, b0, acc0);
    acc1 = wmma_f16(a, b1, acc1);
    __syncthreads();
  }
  int mrow  = bm * 64 + msub * 16 + ((lane >= 16) ? 8 : 0);
  int ncol0 = bn * 64 + (nbase + 0) * 16 + (lane & 15);
  #pragma unroll
  for (int j = 0; j < 8; ++j) {
    int m = mrow + j;
    Cout[(size_t)m * N + ncol0]      = acc0[j];
    Cout[(size_t)m * N + ncol0 + 16] = acc1[j];
  }
}

// ---------------------------------------------------------------------------
// Repack qv1/kv2 (B,200,2,H,D f32) into f16 attention operand layouts.
//   repack_qk: (B,H,208,16) row-padded (s=0 slice)
//   repack_vT: (B,H,16,224) K-padded transpose (s=1 slice)
// ---------------------------------------------------------------------------
__global__ void repack_qk_kernel(const float* __restrict__ src,
                                 _Float16* __restrict__ dst, int soff) {
  int idx = blockIdx.x * 256 + threadIdx.x;
  if (idx >= 128 * 208 * 16) return;
  int d  = idx & 15;
  int rp = (idx >> 4) % 208;
  int bh = idx / (208 * 16);
  int h = bh & 15, b = bh >> 4;
  float v = (rp < 200) ? src[((size_t)(b * 200 + rp)) * 512 + soff + h * 16 + d] : 0.0f;
  dst[idx] = (_Float16)v;
}

__global__ void repack_vT_kernel(const float* __restrict__ src,
                                 _Float16* __restrict__ dst, int soff) {
  int idx = blockIdx.x * 256 + threadIdx.x;
  if (idx >= 128 * 16 * 224) return;
  int cp = idx % 224;
  int d  = (idx / 224) & 15;
  int bh = idx / (224 * 16);
  int h = bh & 15, b = bh >> 4;
  float v = (cp < 200) ? src[((size_t)(b * 200 + cp)) * 512 + soff + h * 16 + d] : 0.0f;
  dst[idx] = (_Float16)v;
}

// ---------------------------------------------------------------------------
// Fused: per (b, 16r x 16c) tile: 16 head dot-WMMAs -> score matrix (256x32)
// in LDS -> MLP 32->512 (relu) ->16 via WMMA, hid chunked 128 cols through LDS.
// Result tile staged back through LDS for coalesced writes of BOTH
// mixed (B,H,200,200) and mixedT (B,H,200,200 transposed r<->c).
// ---------------------------------------------------------------------------
#define NT 13
__global__ void __launch_bounds__(256) mixed_kernel(
    const _Float16* __restrict__ qh, const _Float16* __restrict__ kh,
    const float* __restrict__ cost, const float* __restrict__ alpha,
    const _Float16* __restrict__ W1, const _Float16* __restrict__ W2,
    float* __restrict__ mixed, float* __restrict__ mixedT) {
  __shared__ unsigned int S[256 * 16] __attribute__((aligned(16)));   // 256 x 32 halfs / 4096 f32
  __shared__ _Float16   Hc[256 * 128] __attribute__((aligned(16)));   // hid chunk
  int blk = blockIdx.x;
  int ct = blk % NT, rt = (blk / NT) % NT, b = blk / (NT * NT);
  int t = threadIdx.x, lane = t & 31, w = t >> 5;
  bool hi = lane >= 16;
  int l16 = lane & 15;
  v8h zero8 = {};

  // ---- per-head scaled dot (K=16 zero-padded to 32) -> scores in LDS ----
  #pragma unroll
  for (int hh = 0; hh < 2; ++hh) {
    int h = w * 2 + hh;
    const _Float16* qbase = qh + ((size_t)(b * 16 + h) * 208 + rt * 16) * 16;
    const _Float16* kbase = kh + ((size_t)(b * 16 + h) * 208 + ct * 16) * 16;
    H16 ua, ub;
    ua.h[0] = *(const v8h*)(qbase + l16 * 16 + (hi ? 8 : 0));  // K 0..7 / 8..15
    ua.h[1] = zero8;                                           // K 16..31 pad
    ub.h[0] = *(const v8h*)(kbase + l16 * 16 + (hi ? 8 : 0));  // K 0..7 / 8..15
    ub.h[1] = zero8;                                           // K 16..31 pad
    v8f dt = {};
    dt = wmma_f16(ua.v, ub.v, dt);
    float al = alpha[h];
    #pragma unroll
    for (int j = 0; j < 8; ++j) {
      int rl = j + (hi ? 8 : 0), cl = l16;
      int p = rl * 16 + cl;
      int r = rt * 16 + rl, c = ct * 16 + cl;
      float cst = (r < 200 && c < 200) ? cost[((size_t)b * 200 + r) * 200 + c] : 0.0f;
      _Float16 s0 = (_Float16)(dt[j] * 0.25f);   // NORM = 1/sqrt(16)
      _Float16 s1 = (_Float16)(al * cst);
      S[p * 16 + h] = pack2h(s0, s1);            // scores[p][2h], [2h+1]
    }
  }
  __syncthreads();

  // ---- MLP: mixed(256x16) = relu(S(256x32) @ W1^T) @ W2^T ----
  v8f acc[2] = { {}, {} };
  int mt0 = w * 2;
  const _Float16* Sh = (const _Float16*)S;
  for (int ch = 0; ch < 4; ++ch) {
    // GEMM1 chunk: this wave owns n-tile `w` of the 128-col chunk
    int nglob = ch * 128 + w * 16;
    v16h bm1 = load_b_tile(W1 + (size_t)nglob * 32, 32, lane);
    #pragma unroll
    for (int mt = 0; mt < 16; ++mt) {
      v16h a = load_a_tile(Sh + mt * 16 * 32, 32, lane);
      v8f dd = {};
      dd = wmma_f16(a, bm1, dd);
      #pragma unroll
      for (int j = 0; j < 8; ++j) {
        int p = mt * 16 + j + (hi ? 8 : 0);
        float v = dd[j];
        Hc[p * 128 + w * 16 + l16] = (_Float16)(v > 0.0f ? v : 0.0f);  // relu
      }
    }
    __syncthreads();
    // GEMM2 partial over this 128-wide K chunk; wave owns 2 m-tiles
    #pragma unroll
    for (int mi = 0; mi < 2; ++mi) {
      #pragma unroll
      for (int ks = 0; ks < 4; ++ks) {
        v16h a  = load_a_tile(Hc + (mt0 + mi) * 16 * 128 + ks * 32, 128, lane);
        v16h bm = load_b_tile(W2 + ch * 128 + ks * 32, 512, lane);
        acc[mi] = wmma_f16(a, bm, acc[mi]);
      }
    }
    __syncthreads();
  }

  // ---- stage result tile in LDS (S reused as 16h x 256p f32) ----
  #pragma unroll
  for (int mi = 0; mi < 2; ++mi) {
    #pragma unroll
    for (int j = 0; j < 8; ++j) {
      int p = (mt0 + mi) * 16 + j + (hi ? 8 : 0);
      FU fu; fu.f = acc[mi][j];
      S[l16 * 256 + p] = fu.u;     // [h][p], p = rl*16+cl
    }
  }
  __syncthreads();

  // ---- coalesced writes: mixed[b,h,r,c] and mixedT[b,h,c,r] ----
  #pragma unroll
  for (int i = 0; i < 16; ++i) {
    int idx = i * 256 + t;
    int h = idx >> 8, p = idx & 255;
    int rl = p >> 4, cl = p & 15;
    int r = rt * 16 + rl, c = ct * 16 + cl;
    size_t bh = (size_t)b * 16 + h;
    if (r < 200 && c < 200) {
      FU fu;
      fu.u = S[h * 256 + p];                          // (rl, cl)
      mixed[(bh * 200 + r) * 200 + c] = fu.f;
      fu.u = S[h * 256 + cl * 16 + rl];               // transposed read in LDS
      mixedT[(bh * 200 + c) * 200 + r] = fu.f;
    }
  }
}

// ---------------------------------------------------------------------------
// Masked tanh-softmax, one wave per (b,h,row), row-major streaming input.
// Writes f16 weights zero-padded to 224 in K. Mask strides (mrs,mks) select
// direct (200,1) vs transposed (1,200) mask traversal.
// ---------------------------------------------------------------------------
__global__ void __launch_bounds__(256) softmax_kernel(
    const float* __restrict__ logits, const unsigned char* __restrict__ mask,
    _Float16* __restrict__ out, int mrs, int mks) {
  int gw = blockIdx.x * 8 + (threadIdx.x >> 5);
  int lane = threadIdx.x & 31;
  int row = gw % 200, h = (gw / 200) & 15, b = gw / 3200;
  const float* mp = logits + ((size_t)(b * 16 + h) * 200 + row) * 200;
  const unsigned char* kp = mask + (size_t)b * 200 * 200;
  float e[7];
  float sum = 0.0f;
  #pragma unroll
  for (int i = 0; i < 7; ++i) {
    int c = lane + i * 32;
    float v = 0.0f;
    if (c < 200) {
      float x = tanhf(mp[c]) * 10.0f;
      v = kp[(size_t)row * mrs + (size_t)c * mks] ? 0.0f : __expf(x);
    }
    e[i] = v;
    sum += v;
  }
  #pragma unroll
  for (int off = 16; off > 0; off >>= 1) sum += __shfl_xor(sum, off, 32);
  float inv = (sum > 0.0f) ? 1.0f / sum : 0.0f;  // all-masked -> zeros
  _Float16* op = out + ((size_t)(b * 16 + h) * 208 + row) * 224;
  #pragma unroll
  for (int i = 0; i < 7; ++i) op[lane + i * 32] = (_Float16)(e[i] * inv);
}

// ---------------------------------------------------------------------------
// Combine: ctx(b,row,(h d)) = W(b,h,row,224) @ V^T(b,h,16,224)^T
// 8 waves per block; each wave: one (b,h,rtile) -> 7 WMMAs over K=224.
// ---------------------------------------------------------------------------
__global__ void __launch_bounds__(256) combine_kernel(
    const _Float16* __restrict__ Wt, const _Float16* __restrict__ Vt,
    float* __restrict__ ctx) {
  int gw = blockIdx.x * 8 + (threadIdx.x >> 5);
  if (gw >= 128 * 13) return;                 // wave-uniform exit
  int rt = gw % 13, bh = gw / 13;
  int h = bh & 15, b = bh >> 4;
  int lane = threadIdx.x & 31;
  bool hi = lane >= 16;
  const _Float16* Abase = Wt + ((size_t)bh * 208 + rt * 16) * 224;
  const _Float16* Bbase = Vt + (size_t)bh * 16 * 224;
  v8f acc = {};
  #pragma unroll
  for (int k0 = 0; k0 < 224; k0 += 32) {
    v16h a  = load_a_tile(Abase + k0, 224, lane);
    v16h bm = load_b_tile(Bbase + k0, 224, lane);
    acc = wmma_f16(a, bm, acc);
  }
  #pragma unroll
  for (int j = 0; j < 8; ++j) {
    int r = rt * 16 + j + (hi ? 8 : 0);
    if (r < 200)
      ctx[((size_t)b * 200 + r) * 256 + h * 16 + (lane & 15)] = acc[j];
  }
}

// ---------------------------------------------------------------------------
extern "C" void kernel_launch(void* const* d_in, const int* in_sizes, int n_in,
                              void* d_out, int out_size, void* d_ws, size_t ws_size,
                              hipStream_t stream) {
  (void)in_sizes; (void)n_in; (void)out_size; (void)ws_size;
  const float*         x1    = (const float*)d_in[0];
  const float*         x2    = (const float*)d_in[1];
  const unsigned char* mask  = (const unsigned char*)d_in[2];
  const float*         cost  = (const float*)d_in[3];
  const float*         Wqv1  = (const float*)d_in[4];
  const float*         Wkv2  = (const float*)d_in[5];
  const float*         lin1  = (const float*)d_in[6];
  const float*         lin2  = (const float*)d_in[7];
  const float*         alpha = (const float*)d_in[8];
  const float*         out1  = (const float*)d_in[9];
  const float*         out2  = (const float*)d_in[10];
  float* outp = (float*)d_out;

  char* ws = (char*)d_ws;
  size_t off = 0;
  auto alloc = [&](size_t bytes) -> char* {
    char* p = ws + off;
    off += (bytes + 255) & ~(size_t)255;
    return p;
  };
  _Float16* wqv1h = (_Float16*)alloc(512 * 256 * 2);
  _Float16* wkv2h = (_Float16*)alloc(512 * 256 * 2);
  _Float16* lin1h = (_Float16*)alloc(512 * 32 * 2);
  _Float16* lin2h = (_Float16*)alloc(16 * 512 * 2);
  _Float16* out1h = (_Float16*)alloc(256 * 256 * 2);
  _Float16* out2h = (_Float16*)alloc(256 * 256 * 2);
  float*    qv1f  = (float*)alloc((size_t)1600 * 512 * 4);
  float*    kv2f  = (float*)alloc((size_t)1600 * 512 * 4);
  _Float16* qhp   = (_Float16*)alloc((size_t)128 * 208 * 16 * 2);
  _Float16* khp   = (_Float16*)alloc((size_t)128 * 208 * 16 * 2);
  _Float16* v1t   = (_Float16*)alloc((size_t)128 * 16 * 224 * 2);
  _Float16* v2t   = (_Float16*)alloc((size_t)128 * 16 * 224 * 2);
  float*    mixd  = (float*)alloc((size_t)128 * 200 * 200 * 4);
  float*    mixdT = (float*)alloc((size_t)128 * 200 * 200 * 4);
  _Float16* w1h   = (_Float16*)alloc((size_t)128 * 208 * 224 * 2);
  _Float16* w2h   = (_Float16*)alloc((size_t)128 * 208 * 224 * 2);
  float*    ctx1  = (float*)alloc((size_t)1600 * 256 * 4);
  float*    ctx2  = (float*)alloc((size_t)1600 * 256 * 4);

  // 1) weights -> f16
  cvt_f16_kernel<<<(512 * 256 + 255) / 256, 256, 0, stream>>>(Wqv1, wqv1h, 512 * 256);
  cvt_f16_kernel<<<(512 * 256 + 255) / 256, 256, 0, stream>>>(Wkv2, wkv2h, 512 * 256);
  cvt_f16_kernel<<<(512 * 32 + 255) / 256, 256, 0, stream>>>(lin1, lin1h, 512 * 32);
  cvt_f16_kernel<<<(16 * 512 + 255) / 256, 256, 0, stream>>>(lin2, lin2h, 16 * 512);
  cvt_f16_kernel<<<(256 * 256 + 255) / 256, 256, 0, stream>>>(out1, out1h, 256 * 256);
  cvt_f16_kernel<<<(256 * 256 + 255) / 256, 256, 0, stream>>>(out2, out2h, 256 * 256);

  // 2) projections: (1600x256) @ (512x256)^T
  gemm_f16w_kernel<<<dim3(25, 8), 256, 0, stream>>>(x1, wqv1h, qv1f, 1600, 512, 256);
  gemm_f16w_kernel<<<dim3(25, 8), 256, 0, stream>>>(x2, wkv2h, kv2f, 1600, 512, 256);

  // 3) repack to attention operand layouts
  int nqk = 128 * 208 * 16, nvt = 128 * 16 * 224;
  repack_qk_kernel<<<(nqk + 255) / 256, 256, 0, stream>>>(qv1f, qhp, 0);
  repack_qk_kernel<<<(nqk + 255) / 256, 256, 0, stream>>>(kv2f, khp, 0);
  repack_vT_kernel<<<(nvt + 255) / 256, 256, 0, stream>>>(qv1f, v1t, 256);
  repack_vT_kernel<<<(nvt + 255) / 256, 256, 0, stream>>>(kv2f, v2t, 256);

  // 4) fused dot + mixed-score MLP (writes mixed and mixedT, both coalesced)
  mixed_kernel<<<8 * NT * NT, 256, 0, stream>>>(qhp, khp, cost, alpha, lin1h, lin2h,
                                                mixd, mixdT);

  // 5) masked tanh-softmax both directions, always row-major streaming
  softmax_kernel<<<3200, 256, 0, stream>>>(mixd,  mask, w1h, 200, 1);
  softmax_kernel<<<3200, 256, 0, stream>>>(mixdT, mask, w2h, 1, 200);

  // 6) attention combines (8 waves / block)
  combine_kernel<<<208, 256, 0, stream>>>(w1h, v2t, ctx1);
  combine_kernel<<<208, 256, 0, stream>>>(w2h, v1t, ctx2);

  // 7) output projections -> d_out (h1 then h2)
  gemm_f16w_kernel<<<dim3(25, 4), 256, 0, stream>>>(ctx1, out1h, outp, 1600, 256, 256);
  gemm_f16w_kernel<<<dim3(25, 4), 256, 0, stream>>>(ctx2, out2h, outp + (size_t)1600 * 256,
                                                    1600, 256, 256);
}
